// MultiScaleRetention_50740743635389
// MI455X (gfx1250) — matrix operand order repaired and checked by reading, not compiled, not run on hardware
//
#include <hip/hip_runtime.h>
#include <hip/hip_bf16.h>

// ---------------- problem constants ----------------
constexpr int Bc = 2, Tc = 2048, Dc = 2048, Hc = 8;
constexpr int DK = 256, DV = 512, CHK = 64, NCHK = Tc / CHK;
constexpr int ROWS = Bc * Tc;           // 4096
constexpr int QKC = Hc * DK;            // 2048 (ld for q/k)
constexpr int VC  = Hc * DV;            // 4096 (ld for v/g/o)

typedef __attribute__((ext_vector_type(16))) __bf16 v16bf;
typedef __attribute__((ext_vector_type(8)))  float  v8f;

// ---------------- bf16 <-> f32 helpers (bit-level) --------------------------
__device__ __forceinline__ unsigned short f2bf(float f) {
    union { float f; unsigned u; } v; v.f = f;
    unsigned r = v.u + 0x7FFFu + ((v.u >> 16) & 1u);
    return (unsigned short)(r >> 16);
}
__device__ __forceinline__ float bf2f(unsigned short h) {
    union { unsigned u; float f; } v; v.u = ((unsigned)h) << 16; return v.f;
}

// ---------------- CDNA5 async copy to LDS (ASYNCcnt path) -------------------
// GLOBAL_LOAD_ASYNC_TO_LDS_B128: VDST = per-lane LDS byte address (generic
// pointer low 32 bits == LDS offset per ISA aperture rules), VADDR = 64-bit
// global address, SADDR = off. Tracked by ASYNCcnt.
__device__ __forceinline__ void async_b128(unsigned lds_byte, const void* gaddr) {
    asm volatile("global_load_async_to_lds_b128 %0, %1, off"
                 :: "v"(lds_byte), "v"(gaddr) : "memory");
}
__device__ __forceinline__ void wait_async_le4() {
    asm volatile("s_wait_asynccnt 0x4" ::: "memory");
}
__device__ __forceinline__ void wait_async_0() {
    asm volatile("s_wait_asynccnt 0x0" ::: "memory");
}

// ---------------- WMMA fragment helpers (CDNA5 wave32 layouts) --------------
__device__ __forceinline__ v16bf load_frag_a(const unsigned short* base, int ld) {
    int lane = threadIdx.x & 31;
    const unsigned short* p = base + (size_t)(lane & 15) * ld + ((lane >> 4) << 3);
    union { v16bf v; uint4 q[2]; } u;
    u.q[0] = *reinterpret_cast<const uint4*>(p);
    u.q[1] = *reinterpret_cast<const uint4*>(p + 16);
    return u.v;
}
__device__ __forceinline__ v16bf load_frag_b(const unsigned short* base, int ld) {
    int lane = threadIdx.x & 31;
    const unsigned short* p = base + (size_t)(lane & 15) * ld + ((lane >> 4) << 3);
    union { v16bf v; uint4 q[2]; } u;
    u.q[0] = *reinterpret_cast<const uint4*>(p);
    u.q[1] = *reinterpret_cast<const uint4*>(p + 16);
    return u.v;
}
__device__ __forceinline__ v8f wmma_bf16(v16bf a, v16bf b, v8f c) {
    return __builtin_amdgcn_wmma_f32_16x16x32_bf16(false, a, false, b,
                                                   (short)0, c, false, false);
}
__device__ __forceinline__ void store_c_f32(float* base, int ld, v8f c) {
    int lane = threadIdx.x & 31;
    int col = lane & 15, r0 = (lane >> 4) << 3;
#pragma unroll
    for (int r = 0; r < 8; ++r) base[(size_t)(r0 + r) * ld + col] = c[r];
}
__device__ __forceinline__ void store_c_bf16(unsigned short* base, int ld, v8f c) {
    int lane = threadIdx.x & 31;
    int col = lane & 15, r0 = (lane >> 4) << 3;
#pragma unroll
    for (int r = 0; r < 8; ++r) base[(size_t)(r0 + r) * ld + col] = f2bf(c[r]);
}

// ---------------- elementwise f32 -> bf16 convert ---------------------------
__global__ void cvt_f32_bf16(const float* __restrict__ in,
                             unsigned short* __restrict__ out, int n) {
    int i = blockIdx.x * 256 + threadIdx.x;
    if (i < n) out[i] = f2bf(in[i]);
}

// ---------------- rotary (+optional scale), f32 in -> bf16 out --------------
__global__ void rotary_kernel(const float* __restrict__ Q,
                              unsigned short* __restrict__ out, float scale) {
    int gid = blockIdx.x * 256 + threadIdx.x;      // ROWS*H*128 threads
    int row = gid >> 10;
    int rem = gid & 1023;
    int h = rem >> 7, d = rem & 127;
    int t = row & (Tc - 1);                        // row = b*T + t
    float inv = __expf(-((float)(2 * d) * (1.0f / 256.0f)) * 9.2103403720f);
    float ang = (float)t * inv;
    float s, c; __sincosf(ang, &s, &c);
    size_t idx = (size_t)row * QKC + h * DK + d;
    float x1 = Q[idx], x2 = Q[idx + 128];
    out[idx]       = f2bf((x1 * c - x2 * s) * scale);
    out[idx + 128] = f2bf((x2 * c + x1 * s) * scale);
}

// ---------------- async double-buffered WMMA GEMM ---------------------------
// C[MxN] = A[MxK] * W[NxK]^T. 256 threads = 8 waves (4M x 2N), WG tile 128x128,
// k-step 32. A/B k-tiles staged in LDS via GLOBAL_LOAD_ASYNC_TO_LDS_B128 with
// double buffering; waves read WMMA fragments from LDS (ds_load_b128).
constexpr int LDA = 40;                       // padded LDS leading dim (elems)
constexpr int TILE_E = 128 * LDA;             // elems per tile buffer
constexpr int GEMM_LDS = 4 * TILE_E * 2;      // A0,A1,B0,B1 bytes (40 KB)

template <bool OUT_BF16>
__global__ void gemm_async(const unsigned short* __restrict__ A,
                           const unsigned short* __restrict__ Bm,
                           void* __restrict__ Cv, int M, int N, int K) {
    extern __shared__ char smem_raw[];
    unsigned short* sm = (unsigned short*)smem_raw;
    unsigned smem_base = (unsigned)(size_t)sm;    // LDS byte offset (low 32b)

    int tid = threadIdx.x;
    int w = tid >> 5;
    int wm = w >> 1, wn = w & 1;
    int m0 = blockIdx.y * 128;
    int n0 = blockIdx.x * 128;

    // async stage one 128x32 A tile and one 128x32 B tile into buffer `buf`
    auto issue = [&](int buf, int k0) {
#pragma unroll
        for (int p = 0; p < 2; ++p) {
            int idx = p * 256 + tid;              // 0..511
            int row = idx >> 2, seg = (idx & 3) << 3;   // seg in elems (8)
            const unsigned short* ga = A + (size_t)(m0 + row) * K + k0 + seg;
            unsigned la = smem_base + (unsigned)(buf * TILE_E + row * LDA + seg) * 2u;
            async_b128(la, ga);
        }
#pragma unroll
        for (int p = 0; p < 2; ++p) {
            int idx = p * 256 + tid;
            int row = idx >> 2, seg = (idx & 3) << 3;
            const unsigned short* gb = Bm + (size_t)(n0 + row) * K + k0 + seg;
            unsigned la = smem_base + (unsigned)((2 + buf) * TILE_E + row * LDA + seg) * 2u;
            async_b128(la, gb);
        }
    };

    v8f acc[2][4];
#pragma unroll
    for (int i = 0; i < 2; ++i)
#pragma unroll
        for (int j = 0; j < 4; ++j) acc[i][j] = (v8f){0,0,0,0,0,0,0,0};

    int nk = K >> 5;
    issue(0, 0);
    for (int i = 0; i < nk; ++i) {
        int cur = i & 1;
        if (i + 1 < nk) {
            issue(cur ^ 1, (i + 1) << 5);
            wait_async_le4();                 // current tile's 4 asyncs retired
        } else {
            wait_async_0();
        }
        __syncthreads();                      // publish LDS tile to all waves

        const unsigned short* As = sm + cur * TILE_E;
        const unsigned short* Bs = sm + (2 + cur) * TILE_E;
        v16bf a[2], b[4];
#pragma unroll
        for (int ii = 0; ii < 2; ++ii)
            a[ii] = load_frag_a(As + (wm * 32 + 16 * ii) * LDA, LDA);
#pragma unroll
        for (int j = 0; j < 4; ++j)
            b[j] = load_frag_b(Bs + (wn * 64 + 16 * j) * LDA, LDA);
#pragma unroll
        for (int ii = 0; ii < 2; ++ii)
#pragma unroll
            for (int j = 0; j < 4; ++j)
                acc[ii][j] = wmma_bf16(a[ii], b[j], acc[ii][j]);

        __syncthreads();                      // done reading before overwrite
    }

#pragma unroll
    for (int i = 0; i < 2; ++i)
#pragma unroll
        for (int j = 0; j < 4; ++j) {
            size_t off = (size_t)(m0 + wm * 32 + 16 * i) * N + (n0 + wn * 64 + 16 * j);
            if (OUT_BF16) store_c_bf16((unsigned short*)Cv + off, N, acc[i][j]);
            else          store_c_f32((float*)Cv + off, N, acc[i][j]);
        }
}

// ---------------- chunked retention scan ------------------------------------
// One workgroup per (b,h,slice of 128 DV). S^T slice (128x256 bf16) lives in
// LDS for the entire 32-chunk scan. 8 waves, all matrix math via bf16 WMMA.
__global__ void retention_kernel(const unsigned short* __restrict__ qr,
                                 const unsigned short* __restrict__ kr,
                                 const unsigned short* __restrict__ vb,
                                 float* __restrict__ O) {
    extern __shared__ char smem_raw[];
    unsigned short* St = (unsigned short*)smem_raw;  // 128 x 256  S^T[dv][d]
    unsigned short* kt = St + 128 * 256;             // 256 x 64   (k*kdec)^T[d][m]
    unsigned short* vt = kt + 256 * 64;              // 128 x 64   v^T[dv][m]
    unsigned short* sc = vt + 128 * 64;              //  64 x 64   scores bf16

    int bid = blockIdx.x;
    int slice = bid & 3;
    int h = (bid >> 2) & 7;
    int b = bid >> 5;
    int tid = threadIdx.x;
    int w = tid >> 5, lane = tid & 31;

    float logb = logf(1.0f - exp2f(-5.0f - (float)h));
    float cdec = __expf(logb * 64.0f);

    for (int i = tid; i < 128 * 256; i += 256) St[i] = 0;

    for (int n = 0; n < NCHK; ++n) {
        const unsigned short* qc = qr + (size_t)(b * Tc + n * CHK) * QKC + h * DK;
        const unsigned short* kc = kr + (size_t)(b * Tc + n * CHK) * QKC + h * DK;
        const unsigned short* vc = vb + (size_t)(b * Tc + n * CHK) * VC + h * DV + slice * 128;
        float*                oc = O  + (size_t)(b * Tc + n * CHK) * VC + h * DV + slice * 128;

        __syncthreads();
        for (int idx = tid; idx < 256 * 64; idx += 256) {
            int d = idx >> 6, m = idx & 63;
            float kd = __expf(logb * (float)(63 - m));
            kt[d * 64 + m] = f2bf(bf2f(kc[(size_t)m * QKC + d]) * kd);
        }
        for (int idx = tid; idx < 128 * 64; idx += 256) {
            int dv = idx >> 6, m = idx & 63;
            vt[dv * 64 + m] = vc[(size_t)m * VC + dv];
        }
        __syncthreads();

        // phase 1: scores = q @ k^T, decay-masked
        for (int t = 0; t < 2; ++t) {
            int id = w * 2 + t, ti = id >> 2, tj = id & 3;
            v8f acc = (v8f){0,0,0,0,0,0,0,0};
            for (int k0 = 0; k0 < DK; k0 += 32) {
                v16bf a = load_frag_a(qc + (size_t)(ti * 16) * QKC + k0, QKC);
                v16bf bb = load_frag_b(kc + (size_t)(tj * 16) * QKC + k0, QKC);
                acc = wmma_bf16(a, bb, acc);
            }
            int col = lane & 15, r0 = (lane >> 4) << 3;
#pragma unroll
            for (int r = 0; r < 8; ++r) {
                int i = ti * 16 + r0 + r, j = tj * 16 + col;
                float v = (i >= j) ? acc[r] * __expf(logb * (float)(i - j)) : 0.0f;
                sc[(ti * 16 + r0 + r) * 64 + tj * 16 + col] = f2bf(v);
            }
        }
        __syncthreads();

        // phase 2: o = (q*q_dec) @ S + scores @ v
        for (int t = 0; t < 4; ++t) {
            int id = w * 4 + t, ti = id >> 3, tj = id & 7;
            v8f acc = (v8f){0,0,0,0,0,0,0,0};
            for (int ks = 0; ks < 64; ks += 32) {
                v16bf a = load_frag_a(sc + (ti * 16) * 64 + ks, 64);
                v16bf bb = load_frag_b(vt + (tj * 16) * 64 + ks, 64);
                acc = wmma_bf16(a, bb, acc);
            }
            float qmul = __expf(logb * (float)(ti * 16 + (lane & 15) + 1));
            for (int k0 = 0; k0 < DK; k0 += 32) {
                union { v16bf v; unsigned short s[16]; } u;
                u.v = load_frag_a(qc + (size_t)(ti * 16) * QKC + k0, QKC);
#pragma unroll
                for (int e = 0; e < 16; ++e) u.s[e] = f2bf(bf2f(u.s[e]) * qmul);
                v16bf bb = load_frag_b(St + (tj * 16) * 256 + k0, 256);
                acc = wmma_bf16(u.v, bb, acc);
            }
            store_c_f32(oc + (size_t)(ti * 16) * VC + tj * 16, VC, acc);
        }
        __syncthreads();

        // phase 3: S^T = S^T * chunk_dec + v^T @ (k*kdec)
        for (int t = 0; t < 16; ++t) {
            int id = w * 16 + t, ti = id >> 4, tj = id & 15;
            v8f acc = (v8f){0,0,0,0,0,0,0,0};
            for (int ks = 0; ks < 64; ks += 32) {
                v16bf a = load_frag_a(vt + (ti * 16) * 64 + ks, 64);
                v16bf bb = load_frag_b(kt + (tj * 16) * 64 + ks, 64);
                acc = wmma_bf16(a, bb, acc);
            }
            int col = lane & 15, r0 = (lane >> 4) << 3;
#pragma unroll
            for (int r = 0; r < 8; ++r) {
                unsigned short* p = St + (size_t)(ti * 16 + r0 + r) * 256 + tj * 16 + col;
                *p = f2bf(bf2f(*p) * cdec + acc[r]);
            }
        }
    }
}

// ---------------- gated RMS norm + swish gate (wave32 reduction) ------------
__global__ void gated_norm(const float* __restrict__ O,
                           const unsigned short* __restrict__ G,
                           const float* __restrict__ gnw,
                           unsigned short* __restrict__ OG) {
    int gw = blockIdx.x * 8 + (threadIdx.x >> 5);
    int lane = threadIdx.x & 31;
    int h = gw & 7, row = gw >> 3;
    const float* op = O + (size_t)row * VC + h * DV;
    const unsigned short* gp = G + (size_t)row * VC + h * DV;
    unsigned short* outp = OG + (size_t)row * VC + h * DV;
    float ov[16], ss = 0.0f;
#pragma unroll
    for (int j = 0; j < 16; ++j) { ov[j] = op[lane + j * 32]; ss += ov[j] * ov[j]; }
#pragma unroll
    for (int off = 16; off > 0; off >>= 1) ss += __shfl_xor(ss, off, 32);
    float rms = rsqrtf(ss * (1.0f / 512.0f) + 1e-5f);
#pragma unroll
    for (int j = 0; j < 16; ++j) {
        int dv = lane + j * 32;
        float g = bf2f(gp[dv]);
        float sw = g / (1.0f + __expf(-g));
        outp[dv] = f2bf(ov[j] * rms * gnw[dv] * sw);
    }
}

// ---------------- host-side orchestration -----------------------------------
extern "C" void kernel_launch(void* const* d_in, const int* in_sizes, int n_in,
                              void* d_out, int out_size, void* d_ws, size_t ws_size,
                              hipStream_t stream) {
    const float* x   = (const float*)d_in[0];
    const float* Wq  = (const float*)d_in[1];
    const float* Wk  = (const float*)d_in[2];
    const float* Wv  = (const float*)d_in[3];
    const float* Wg  = (const float*)d_in[4];
    const float* Wo  = (const float*)d_in[5];
    const float* gnw = (const float*)d_in[6];
    float* out = (float*)d_out;

    char* p = (char*)d_ws;
    auto alloc = [&](size_t bytes) -> void* {
        void* r = (void*)p;
        p += (bytes + 255) & ~(size_t)255;
        return r;
    };
    unsigned short* xb  = (unsigned short*)alloc((size_t)ROWS * Dc * 2);
    unsigned short* wqb = (unsigned short*)alloc((size_t)QKC * Dc * 2);
    unsigned short* wkb = (unsigned short*)alloc((size_t)QKC * Dc * 2);
    unsigned short* wvb = (unsigned short*)alloc((size_t)VC * Dc * 2);
    unsigned short* wgb = (unsigned short*)alloc((size_t)VC * Dc * 2);
    unsigned short* wob = (unsigned short*)alloc((size_t)Dc * VC * 2);
    float*          tmpF= (float*)alloc((size_t)ROWS * QKC * 4);
    unsigned short* qrb = (unsigned short*)alloc((size_t)ROWS * QKC * 2);
    unsigned short* krb = (unsigned short*)alloc((size_t)ROWS * QKC * 2);
    unsigned short* vbb = (unsigned short*)alloc((size_t)ROWS * VC * 2);
    unsigned short* gbb = (unsigned short*)alloc((size_t)ROWS * VC * 2);
    float*          Obf = (float*)alloc((size_t)ROWS * VC * 4);
    unsigned short* OGb = (unsigned short*)alloc((size_t)ROWS * VC * 2);

    auto cvt = [&](const float* src, unsigned short* dst, int n) {
        cvt_f32_bf16<<<(n + 255) / 256, 256, 0, stream>>>(src, dst, n);
    };
    cvt(x,  xb,  ROWS * Dc);
    cvt(Wq, wqb, QKC * Dc);
    cvt(Wk, wkb, QKC * Dc);
    cvt(Wv, wvb, VC * Dc);
    cvt(Wg, wgb, VC * Dc);
    cvt(Wo, wob, Dc * VC);

    dim3 blk(256);
    dim3 gQ(QKC / 128, ROWS / 128);   // (16, 32)
    dim3 gV(VC / 128, ROWS / 128);    // (32, 32)

    gemm_async<false><<<gQ, blk, GEMM_LDS, stream>>>(xb, wqb, tmpF, ROWS, QKC, Dc);
    rotary_kernel<<<(ROWS * Hc * 128) / 256, blk, 0, stream>>>(tmpF, qrb, 0.0625f);
    gemm_async<false><<<gQ, blk, GEMM_LDS, stream>>>(xb, wkb, tmpF, ROWS, QKC, Dc);
    rotary_kernel<<<(ROWS * Hc * 128) / 256, blk, 0, stream>>>(tmpF, krb, 1.0f);
    gemm_async<true><<<gV, blk, GEMM_LDS, stream>>>(xb, wvb, vbb, ROWS, VC, Dc);
    gemm_async<true><<<gV, blk, GEMM_LDS, stream>>>(xb, wgb, gbb, ROWS, VC, Dc);

    size_t lds = (size_t)(128 * 256 + 256 * 64 + 128 * 64 + 64 * 64) * 2;
    retention_kernel<<<Bc * Hc * 4, blk, lds, stream>>>(qrb, krb, vbb, Obf);

    gated_norm<<<(ROWS * Hc) / 8, blk, 0, stream>>>(Obf, gbb, gnw, OGb);

    gemm_async<false><<<gQ, blk, GEMM_LDS, stream>>>(OGb, wob, out, ROWS, Dc, VC);
}
